// MultiHeadAttention_39840116638308
// MI455X (gfx1250) — compile-verified, matrix operand order checked by reading, not run on hardware
//
#include <hip/hip_runtime.h>
#include <hip/hip_bf16.h>

#define BB 4
#define SS 1024
#define DD 2048
#define HH 16
#define DHD 128
#define MM (BB*SS)

#define CHUNK_K 128                 // k-elements staged per barrier interval
#define LDS_STRIDE 136              // 128 padded to 136 -> conflict-free ds reads

typedef __bf16 bf16;
typedef __attribute__((ext_vector_type(16))) __bf16 v16bf;
typedef __attribute__((ext_vector_type(8)))  __bf16 v8bf;
typedef __attribute__((ext_vector_type(4)))  __bf16 v4bf;
typedef __attribute__((ext_vector_type(8)))  float  v8f;
typedef __attribute__((ext_vector_type(4)))  float  v4f;

// Workgroup barrier with LDS-only fence: drains DScnt (so staged LDS data is
// visible to the whole WGP) but leaves global loads in flight across the
// barrier -- the minimal CDNA5 split-barrier sequence.
__device__ __forceinline__ void wg_barrier_lds(void) {
    asm volatile("s_wait_dscnt 0x0\n\t"
                 "s_barrier_signal -1\n\t"
                 "s_barrier_wait -1" ::: "memory");
}

// ---- WMMA fragment helpers (16x32 A / 32x16 B from an N-major, K-contiguous matrix) ----
// Per ISA 7.12.2: lane<16 holds row=lane, k = 0..7 (v0-3) and 16..23 (v4-7);
// lane>=16 holds row=lane-16, k = 8..15 and 24..31.
__device__ __forceinline__ v16bf frag_pack(v8bf lo, v8bf hi) {
    v16bf f;
#pragma unroll
    for (int i = 0; i < 8; ++i) { f[i] = lo[i]; f[i + 8] = hi[i]; }
    return f;
}

__device__ __forceinline__ v16bf frag_ld_bf16(const bf16* row0, int ld, int k0) {
    const int lane = threadIdx.x & 31;
    const int half = lane >> 4, r = lane & 15;
    const bf16* p = row0 + (size_t)r * ld + k0 + half * 8;
    return frag_pack(*reinterpret_cast<const v8bf*>(p),
                     *reinterpret_cast<const v8bf*>(p + 16));
}

__device__ __forceinline__ v16bf frag_ld_f32(const float* row0, int ld, int k0) {
    const int lane = threadIdx.x & 31;
    const int half = lane >> 4, r = lane & 15;
    const float* p = row0 + (size_t)r * ld + k0 + half * 8;
    v4f a0 = *reinterpret_cast<const v4f*>(p);
    v4f a1 = *reinterpret_cast<const v4f*>(p + 4);
    v4f b0 = *reinterpret_cast<const v4f*>(p + 16);
    v4f b1 = *reinterpret_cast<const v4f*>(p + 20);
    v16bf f;
#pragma unroll
    for (int i = 0; i < 4; ++i) {
        f[i]      = (bf16)a0[i];
        f[i + 4]  = (bf16)a1[i];
        f[i + 8]  = (bf16)b0[i];
        f[i + 12] = (bf16)b1[i];
    }
    return f;
}

// ---- f32 -> bf16 bulk convert ----
__global__ __launch_bounds__(256) void cvt_f32_bf16(const float* __restrict__ in,
                                                    bf16* __restrict__ out, int n) {
    int i = (blockIdx.x * 256 + threadIdx.x) * 4;
    if (i + 3 < n) {
        v4f v = *reinterpret_cast<const v4f*>(in + i);
        v4bf o;
#pragma unroll
        for (int j = 0; j < 4; ++j) o[j] = (bf16)v[j];
        *reinterpret_cast<v4bf*>(out + i) = o;
    }
}

// ---- NT GEMM: C[M,N] = A[M,K] * B[N,K]^T, bf16 in, f32 accum, f32 or bf16 out ----
// block = 128 (4 waves). wave w -> rows m0 = by*64 + w*16; cols n0 = bx*64.
// B panel staged chunk-by-chunk (64n x 128k = 16KB) into double-buffered LDS:
// one LDS-fence barrier per chunk; next chunk's B stage + A fragments are
// global-prefetched into registers during the current chunk's 16 WMMAs.
__global__ __launch_bounds__(128) void gemm_nt(const bf16* __restrict__ A,
                                               const bf16* __restrict__ Bm,
                                               float* __restrict__ Cf,
                                               bf16* __restrict__ Cb,
                                               int Ndim, int Kdim) {
    __shared__ bf16 Bs[2][64 * LDS_STRIDE];   // 2 x 17KB

    const int tid  = threadIdx.x;
    const int wave = tid >> 5;
    const int lane = tid & 31;
    const int m0 = blockIdx.y * 64 + wave * 16;
    const int n0 = blockIdx.x * 64;

    // cooperative staging: thread owns B row (tid>>1), 64-element k-segment (tid&1)*64
    const int srow = tid >> 1;
    const int scol = (tid & 1) * 64;
    const bf16* Bsrc = Bm + (size_t)(n0 + srow) * Kdim + scol;
    const int  sdst  = srow * LDS_STRIDE + scol;

    // A fragment base for this lane (ISA 7.12.2 layout)
    const int ahalf = lane >> 4, ar = lane & 15;
    const bf16* Ap = A + (size_t)(m0 + ar) * Kdim + ahalf * 8;

    v8f acc[4];
#pragma unroll
    for (int t = 0; t < 4; ++t)
#pragma unroll
        for (int i = 0; i < 8; ++i) acc[t][i] = 0.0f;

    const int nCh = Kdim / CHUNK_K;

    // ---- prologue: fetch + stage chunk 0 ----
    v8bf stg[8];
#pragma unroll
    for (int j = 0; j < 8; ++j)
        stg[j] = *reinterpret_cast<const v8bf*>(Bsrc + j * 8);
    v8bf aclo[4], achi[4];
#pragma unroll
    for (int s = 0; s < 4; ++s) {
        aclo[s] = *reinterpret_cast<const v8bf*>(Ap + s * 32);
        achi[s] = *reinterpret_cast<const v8bf*>(Ap + s * 32 + 16);
    }
#pragma unroll
    for (int j = 0; j < 8; ++j)
        *reinterpret_cast<v8bf*>(&Bs[0][sdst + j * 8]) = stg[j];
    wg_barrier_lds();

    for (int c = 0; c < nCh; ++c) {
        const int kn = (c + 1) * CHUNK_K;     // base of next chunk
        v8bf anlo[4], anhi[4];
        if (c + 1 < nCh) {
            // global prefetch of next chunk (kept in flight across this chunk's WMMAs)
#pragma unroll
            for (int j = 0; j < 8; ++j)
                stg[j] = *reinterpret_cast<const v8bf*>(Bsrc + kn + j * 8);
#pragma unroll
            for (int s = 0; s < 4; ++s) {
                anlo[s] = *reinterpret_cast<const v8bf*>(Ap + kn + s * 32);
                anhi[s] = *reinterpret_cast<const v8bf*>(Ap + kn + s * 32 + 16);
            }
        }

        // ---- compute: 4 k-steps x 4 n-tiles = 16 WMMAs from LDS buffer c&1 ----
        const bf16* base = &Bs[c & 1][(lane & 15) * LDS_STRIDE + (lane >> 4) * 8];
#pragma unroll
        for (int s = 0; s < 4; ++s) {
            v16bf af = frag_pack(aclo[s], achi[s]);
            v16bf bfr[4];
#pragma unroll
            for (int t = 0; t < 4; ++t) {
                const bf16* bp = base + t * 16 * LDS_STRIDE + s * 32;
                bfr[t] = frag_pack(*reinterpret_cast<const v8bf*>(bp),
                                   *reinterpret_cast<const v8bf*>(bp + 16));
            }
#pragma unroll
            for (int t = 0; t < 4; ++t)
                acc[t] = __builtin_amdgcn_wmma_f32_16x16x32_bf16(
                    false, af, false, bfr[t], (short)0, acc[t], false, false);
        }

        if (c + 1 < nCh) {
            // stage next chunk into the other LDS buffer; waits only on its own loads
#pragma unroll
            for (int j = 0; j < 8; ++j)
                *reinterpret_cast<v8bf*>(&Bs[(c + 1) & 1][sdst + j * 8]) = stg[j];
#pragma unroll
            for (int s = 0; s < 4; ++s) { aclo[s] = anlo[s]; achi[s] = anhi[s]; }
        }
        wg_barrier_lds();
    }

    const int mb = (lane >> 4) * 8, nn = lane & 15;
#pragma unroll
    for (int t = 0; t < 4; ++t)
#pragma unroll
        for (int r = 0; r < 8; ++r) {
            const size_t idx = (size_t)(m0 + mb + r) * Ndim + n0 + t * 16 + nn;
            if (Cf) Cf[idx] = acc[t][r];
            else    Cb[idx] = (bf16)acc[t][r];
        }
}

// ---- Llama half-split RoPE, in place on bf16 [B,S,H*128]; one thread per (b,s,h,j<64) pair ----
__global__ __launch_bounds__(256) void rope_inplace(bf16* __restrict__ qk) {
    int idx = blockIdx.x * 256 + threadIdx.x;
    if (idx >= BB * SS * HH * 64) return;
    const int j = idx & 63;
    const int h = (idx >> 6) & (HH - 1);
    const int s = (idx >> 10) & (SS - 1);
    const int b = idx >> 20;
    const float inv = __expf(-0.14391156856f * (float)j);   // 10000^(-2j/128)
    const float fr = (float)s * inv;
    const float c = __cosf(fr), sn = __sinf(fr);
    const size_t base = (size_t)(b * SS + s) * DD + h * DHD + j;
    const float x1 = (float)qk[base], x2 = (float)qk[base + 64];
    qk[base]      = (bf16)(x1 * c - x2 * sn);
    qk[base + 64] = (bf16)(x2 * c + x1 * sn);
}

// ---- per-head transpose: Vt[bh][dv][s] = V[b][s][h*128+dv] ----
__global__ __launch_bounds__(256) void transpose_v(const bf16* __restrict__ Vb,
                                                   bf16* __restrict__ Vt) {
    int idx = blockIdx.x * 256 + threadIdx.x;
    if (idx >= BB * HH * DHD * SS) return;
    const int s  = idx & (SS - 1);
    const int dv = (idx >> 10) & (DHD - 1);
    const int h  = (idx >> 17) & (HH - 1);
    const int b  = idx >> 21;
    Vt[idx] = Vb[(size_t)(b * SS + s) * DD + h * DHD + dv];
}

// ---- scores = Q K^T / sqrt(d) + causal + mask -> f32 attn slice of d_out (HBM-bound) ----
__global__ __launch_bounds__(128) void attn_scores(const bf16* __restrict__ Q,
                                                   const bf16* __restrict__ Kb,
                                                   const float* __restrict__ mask,
                                                   float* __restrict__ attnW) {
    const int bh = blockIdx.z;
    const int b = bh >> 4, h = bh & 15;
    const int wave = threadIdx.x >> 5;
    const int q0 = blockIdx.y * 64 + wave * 16;
    const int n0 = blockIdx.x * 64;
    const bf16* Qp = Q  + (size_t)b * SS * DD + h * DHD;   // row stride DD
    const bf16* Kp = Kb + (size_t)b * SS * DD + h * DHD;
    v8f acc[4];
#pragma unroll
    for (int t = 0; t < 4; ++t)
#pragma unroll
        for (int i = 0; i < 8; ++i) acc[t][i] = 0.0f;

#pragma unroll
    for (int k0 = 0; k0 < DHD; k0 += 32) {
        v16bf af = frag_ld_bf16(Qp + (size_t)q0 * DD, DD, k0);
#pragma unroll
        for (int t = 0; t < 4; ++t) {
            v16bf bf = frag_ld_bf16(Kp + (size_t)(n0 + t * 16) * DD, DD, k0);
            acc[t] = __builtin_amdgcn_wmma_f32_16x16x32_bf16(
                false, af, false, bf, (short)0, acc[t], false, false);
        }
    }
    const int lane = threadIdx.x & 31;
    const int mb = (lane >> 4) * 8, nn = lane & 15;
    const float scale = 0.08838834764831845f;  // 1/sqrt(128)
#pragma unroll
    for (int t = 0; t < 4; ++t)
#pragma unroll
        for (int r = 0; r < 8; ++r) {
            const int qr = q0 + mb + r, kc = n0 + t * 16 + nn;
            float sc = acc[t][r] * scale + mask[b * SS + kc];
            if (kc > qr) sc = -__builtin_inff();
            attnW[((size_t)bh * SS + qr) * SS + kc] = sc;
        }
}

// ---- softmax over rows of 1024; one wave per row, row held in 32 regs/lane ----
__global__ __launch_bounds__(256) void softmax_rows(float* __restrict__ attnW) {
    const int row = blockIdx.x * 8 + (threadIdx.x >> 5);
    const int lane = threadIdx.x & 31;
    float* p = attnW + (size_t)row * SS;
    float v[32];
    float mx = -__builtin_inff();
#pragma unroll
    for (int i = 0; i < 32; ++i) { v[i] = p[i * 32 + lane]; mx = fmaxf(mx, v[i]); }
#pragma unroll
    for (int off = 16; off >= 1; off >>= 1) mx = fmaxf(mx, __shfl_xor(mx, off, 32));
    float sum = 0.0f;
#pragma unroll
    for (int i = 0; i < 32; ++i) { v[i] = __expf(v[i] - mx); sum += v[i]; }
#pragma unroll
    for (int off = 16; off >= 1; off >>= 1) sum += __shfl_xor(sum, off, 32);
    const float rs = 1.0f / sum;
#pragma unroll
    for (int i = 0; i < 32; ++i) p[i * 32 + lane] = v[i] * rs;
}

// ---- ctx = attn @ V   (A = f32 attn converted on load, B = per-head-transposed V) ----
__global__ __launch_bounds__(128) void attn_av(const float* __restrict__ attnW,
                                               const bf16* __restrict__ Vt,
                                               bf16* __restrict__ ctx) {
    const int bh = blockIdx.z;
    const int b = bh >> 4, h = bh & 15;
    const int wave = threadIdx.x >> 5;
    const int q0  = blockIdx.y * 32 + (wave >> 1) * 16;
    const int dv0 = (wave & 1) * 64;
    const float* Ap = attnW + (size_t)bh * SS * SS;
    const bf16*  Vp = Vt + (size_t)bh * DHD * SS;          // row dv, stride SS
    v8f acc[4];
#pragma unroll
    for (int t = 0; t < 4; ++t)
#pragma unroll
        for (int i = 0; i < 8; ++i) acc[t][i] = 0.0f;

    for (int k0 = 0; k0 < SS; k0 += 32) {
        v16bf af = frag_ld_f32(Ap + (size_t)q0 * SS, SS, k0);
#pragma unroll
        for (int t = 0; t < 4; ++t) {
            v16bf bf = frag_ld_bf16(Vp + (size_t)(dv0 + t * 16) * SS, SS, k0);
            acc[t] = __builtin_amdgcn_wmma_f32_16x16x32_bf16(
                false, af, false, bf, (short)0, acc[t], false, false);
        }
    }
    const int lane = threadIdx.x & 31;
    const int mb = (lane >> 4) * 8, nn = lane & 15;
#pragma unroll
    for (int t = 0; t < 4; ++t)
#pragma unroll
        for (int r = 0; r < 8; ++r)
            ctx[(size_t)(b * SS + q0 + mb + r) * DD + h * DHD + dv0 + t * 16 + nn] =
                (bf16)acc[t][r];
}

extern "C" void kernel_launch(void* const* d_in, const int* in_sizes, int n_in,
                              void* d_out, int out_size, void* d_ws, size_t ws_size,
                              hipStream_t stream) {
    const float* x    = (const float*)d_in[0];
    const float* mask = (const float*)d_in[1];
    const float* Wq   = (const float*)d_in[2];
    const float* Wk   = (const float*)d_in[3];
    const float* Wv   = (const float*)d_in[4];
    const float* Wo   = (const float*)d_in[5];

    float* out   = (float*)d_out;
    float* attnW = out + (size_t)BB * SS * DD;   // second tuple output

    char* ws = (char*)d_ws;
    size_t off = 0;
    auto alloc = [&](size_t bytes) {
        char* p = ws + off;
        off += (bytes + 255) & ~(size_t)255;
        return p;
    };
    bf16* xb  = (bf16*)alloc((size_t)MM * DD * 2);      // 16.8 MB
    bf16* Wqb = (bf16*)alloc((size_t)DD * DD * 2);
    bf16* Wkb = (bf16*)alloc((size_t)DD * DD * 2);
    bf16* Wvb = (bf16*)alloc((size_t)DD * DD * 2);
    bf16* Wob = (bf16*)alloc((size_t)DD * DD * 2);      // +33.6 MB
    bf16* Qb  = (bf16*)alloc((size_t)MM * DD * 2);
    bf16* Kb  = (bf16*)alloc((size_t)MM * DD * 2);
    bf16* Vb  = (bf16*)alloc((size_t)MM * DD * 2);      // +50.3 MB
    bf16* Vt  = (bf16*)alloc((size_t)BB * HH * DHD * SS * 2); // +16.8 MB => ~117.4 MB
    bf16* ctxb = xb;   // x dead after projections; reuse its buffer for context

    // 1) downconvert activations + weights to bf16
    cvt_f32_bf16<<<(MM * DD / 4 + 255) / 256, 256, 0, stream>>>(x, xb, MM * DD);
    cvt_f32_bf16<<<(DD * DD / 4 + 255) / 256, 256, 0, stream>>>(Wq, Wqb, DD * DD);
    cvt_f32_bf16<<<(DD * DD / 4 + 255) / 256, 256, 0, stream>>>(Wk, Wkb, DD * DD);
    cvt_f32_bf16<<<(DD * DD / 4 + 255) / 256, 256, 0, stream>>>(Wv, Wvb, DD * DD);
    cvt_f32_bf16<<<(DD * DD / 4 + 255) / 256, 256, 0, stream>>>(Wo, Wob, DD * DD);

    // 2) Q/K/V projections (x @ W^T), bf16 out
    dim3 gGemm(DD / 64, MM / 64);
    gemm_nt<<<gGemm, 128, 0, stream>>>(xb, Wqb, nullptr, Qb, DD, DD);
    gemm_nt<<<gGemm, 128, 0, stream>>>(xb, Wkb, nullptr, Kb, DD, DD);
    gemm_nt<<<gGemm, 128, 0, stream>>>(xb, Wvb, nullptr, Vb, DD, DD);

    // 3) RoPE on Q and K (in place)
    rope_inplace<<<(BB * SS * HH * 64 + 255) / 256, 256, 0, stream>>>(Qb);
    rope_inplace<<<(BB * SS * HH * 64 + 255) / 256, 256, 0, stream>>>(Kb);

    // 4) per-head V transpose for N-major WMMA B fragments
    transpose_v<<<(BB * HH * DHD * SS + 255) / 256, 256, 0, stream>>>(Vb, Vt);

    // 5) scores -> attn slice of d_out (f32)
    attn_scores<<<dim3(SS / 64, SS / 64, BB * HH), 128, 0, stream>>>(Qb, Kb, mask, attnW);

    // 6) softmax rows in place
    softmax_rows<<<BB * HH * SS / 8, 256, 0, stream>>>(attnW);

    // 7) ctx = attn @ V (bf16 out)
    attn_av<<<dim3(1, SS / 32, BB * HH), 128, 0, stream>>>(attnW, Vt, ctxb);

    // 8) out = ctx @ Wo^T (f32 out)
    gemm_nt<<<gGemm, 128, 0, stream>>>(ctxb, Wob, out, nullptr, DD, DD);
}
